// OrbNet_9131100471468
// MI455X (gfx1250) — matrix operand order repaired and checked by reading, not compiled, not run on hardware
//
#include <hip/hip_runtime.h>
#include <hip/hip_bf16.h>
#include <math.h>

// OrbNet-style GNN forward for MI455X (gfx1250, wave32).
// All GEMMs run through v_wmma_f32_16x16x32_f16; edge tensors kept in f16
// (HBM-bound workload); f32 only for BN stats, scatter accumulator, output.

typedef __attribute__((ext_vector_type(16))) _Float16 v16h;
typedef __attribute__((ext_vector_type(8)))  _Float16 v8h;
typedef __attribute__((ext_vector_type(8)))  float    v8f;

#define NNODES 50000
#define NEDGES 800000

__device__ __forceinline__ float siluf(float x) { return x / (1.0f + __expf(-x)); }

// ---------------------------------------------------------------------------
// Generic WMMA GEMM:  C[M, Nc] = act( A[M, Kpad(f16)] @ W[Nc, K(f32)]^T + b )
// Block = 128 threads (4 waves). Each wave computes one 16x64 tile.
// grid.x = ceil(M/64), grid.y = Nc/64. Weight panel (64 x Kpad f16) in LDS.
// ---------------------------------------------------------------------------
__global__ __launch_bounds__(128) void k_gemm(
    const _Float16* __restrict__ A, int lda,
    const float* __restrict__ W, int K, int Kpad,
    const float* __restrict__ bias,
    _Float16* __restrict__ C, int ldc, int M, int act)
{
  __shared__ _Float16 __attribute__((aligned(32))) Wl[64 * 288];
  const int n0  = blockIdx.y * 64;
  const int tid = threadIdx.x;
  const int total = 64 * Kpad;
  for (int idx = tid; idx < total; idx += 128) {
    int n = idx / Kpad;
    int k = idx - n * Kpad;
    float w = (k < K) ? W[(size_t)(n0 + n) * K + k] : 0.0f;
    Wl[n * Kpad + k] = (_Float16)w;
  }
  __syncthreads();

  const int lane = tid & 31;
  const int m0   = blockIdx.x * 64 + (tid >> 5) * 16;
  if (m0 >= M) return;               // M is always a multiple of 16
  const int hi = lane >> 4;          // 0: K-group low, 1: K-group high
  const int lm = lane & 15;

  const _Float16* Arow = A + (size_t)(m0 + lm) * lda;
  v8f acc0 = {0,0,0,0,0,0,0,0};
  v8f acc1 = acc0, acc2 = acc0, acc3 = acc0;

  for (int kk = 0; kk < Kpad; kk += 32) {
    // A fragment, contiguous K order matching B:
    // lanes 0-15: K = kk+0..15; lanes 16-31: K = kk+16..31
    v16h a = *(const v16h*)(Arow + kk + hi * 16);

    // B fragment (W^T): lane<16 -> K=kk..kk+15 of column lm; lane>=16 -> kk+16..kk+31
    const int ko = kk + hi * 16;
    v16h b0 = *(const v16h*)&Wl[(lm     ) * Kpad + ko];
    v16h b1 = *(const v16h*)&Wl[(lm + 16) * Kpad + ko];
    v16h b2 = *(const v16h*)&Wl[(lm + 32) * Kpad + ko];
    v16h b3 = *(const v16h*)&Wl[(lm + 48) * Kpad + ko];

    acc0 = __builtin_amdgcn_wmma_f32_16x16x32_f16(false, a, false, b0, (short)0, acc0, false, false);
    acc1 = __builtin_amdgcn_wmma_f32_16x16x32_f16(false, a, false, b1, (short)0, acc1, false, false);
    acc2 = __builtin_amdgcn_wmma_f32_16x16x32_f16(false, a, false, b2, (short)0, acc2, false, false);
    acc3 = __builtin_amdgcn_wmma_f32_16x16x32_f16(false, a, false, b3, (short)0, acc3, false, false);
  }

  const int c0 = n0 + lm;
  float bb0 = bias ? bias[c0     ] : 0.0f;
  float bb1 = bias ? bias[c0 + 16] : 0.0f;
  float bb2 = bias ? bias[c0 + 32] : 0.0f;
  float bb3 = bias ? bias[c0 + 48] : 0.0f;
#pragma unroll
  for (int g = 0; g < 8; ++g) {
    const size_t row = (size_t)(m0 + g + hi * 8);
    float v0 = acc0[g] + bb0, v1 = acc1[g] + bb1;
    float v2 = acc2[g] + bb2, v3 = acc3[g] + bb3;
    if (act == 1) { v0 = siluf(v0); v1 = siluf(v1); v2 = siluf(v2); v3 = siluf(v3); }
    _Float16* Crow = C + row * ldc + c0;
    Crow[0]  = (_Float16)v0; Crow[16] = (_Float16)v1;
    Crow[32] = (_Float16)v2; Crow[48] = (_Float16)v3;
  }
}

// ---------------------------------------------------------------------------
// Elementwise / reduction helpers
// ---------------------------------------------------------------------------
__global__ void k_zero_f32(float* p, long n) {
  long i = blockIdx.x * (long)blockDim.x + threadIdx.x;
  if (i < n) p[i] = 0.0f;
}

__global__ void k_rbf(const float* __restrict__ d, _Float16* __restrict__ out, int E) {
  int e = blockIdx.x * blockDim.x + threadIdx.x;
  if (e >= E) return;
  float x = d[e];
  const float step  = 6.0f / 7.0f;
  const float coeff = -0.5f / (step * step);
  float xs = x * (1.0f / 6.0f);
  float x2 = xs * xs;
  float x5 = x2 * x2 * xs;
  float env = 1.0f - 21.0f * x5 + 35.0f * x5 * xs - 15.0f * x5 * x2;
  if (xs >= 1.0f) env = 0.0f;
  size_t b = (size_t)e * 32;
#pragma unroll
  for (int i = 0; i < 8; ++i) {
    float off = step * (float)i;
    float g = __expf(coeff * (x - off) * (x - off));
    out[b + i] = (_Float16)(g * env);
  }
#pragma unroll
  for (int i = 8; i < 32; ++i) out[b + i] = (_Float16)0.0f;
}

__global__ void k_embed(const int* __restrict__ period, const int* __restrict__ group,
                        const float* __restrict__ nbf, const float* __restrict__ embP,
                        const float* __restrict__ embG, _Float16* __restrict__ out, long total) {
  long i = blockIdx.x * (long)blockDim.x + threadIdx.x;
  if (i >= total) return;
  int n = (int)(i / 288);
  int c = (int)(i - (long)n * 288);
  float v;
  if (c < 128)      v = embP[period[n] * 128 + c];
  else if (c < 256) v = embG[group[n] * 128 + (c - 128)];
  else if (c < 260) v = nbf[n * 4 + (c - 256)];
  else              v = 0.0f;
  out[i] = (_Float16)v;
}

__global__ void k_add16(const _Float16* a, const _Float16* b, _Float16* o, long n) {
  long i = blockIdx.x * (long)blockDim.x + threadIdx.x;
  if (i < n) o[i] = (_Float16)((float)a[i] + (float)b[i]);
}

__global__ void k_hinit(const _Float16* a, const _Float16* b, float* h32, _Float16* h16, long n) {
  long i = blockIdx.x * (long)blockDim.x + threadIdx.x;
  if (i < n) { float v = (float)a[i] + (float)b[i]; h32[i] = v; h16[i] = (_Float16)v; }
}

__global__ void k_hupdate(const _Float16* a, float* h32, _Float16* h16, long n) {
  long i = blockIdx.x * (long)blockDim.x + threadIdx.x;
  if (i < n) { float v = h32[i] + (float)a[i]; h32[i] = v; h16[i] = (_Float16)v; }
}

__global__ void k_f16cvt(const float* a, _Float16* o, long n) {
  long i = blockIdx.x * (long)blockDim.x + threadIdx.x;
  if (i < n) o[i] = (_Float16)a[i];
}

// column sums / sumsq for BatchNorm (training-mode stats). blockDim == C.
__global__ void k_colstats(const _Float16* __restrict__ X, float* __restrict__ stats, int M, int C) {
  int c = threadIdx.x;
  float s = 0.0f, q = 0.0f;
  for (int r = blockIdx.x; r < M; r += gridDim.x) {
    float v = (float)X[(size_t)r * C + c];
    s += v; q += v * v;
  }
  atomicAdd(&stats[c], s);
  atomicAdd(&stats[C + c], q);
}

__global__ void k_bnsilu(const _Float16* __restrict__ X, const float* __restrict__ stats,
                         const float* __restrict__ g, const float* __restrict__ be,
                         _Float16* __restrict__ Y, long total, int C, float invM) {
  long i = blockIdx.x * (long)blockDim.x + threadIdx.x;
  if (i >= total) return;
  int c = (int)(i % C);
  float mu  = stats[c] * invM;
  float var = stats[C + c] * invM - mu * mu;
  float x = (float)X[i];
  float y = g[c] * (x - mu) * rsqrtf(var + 1e-5f) + be[c];
  Y[i] = (_Float16)siluf(y);
}

// prod[e,c] = x_down[src[e],c] * x_down[dst[e],c] * e[e,c]
__global__ void k_prod(const _Float16* __restrict__ xd, const _Float16* __restrict__ e,
                       const int* __restrict__ ei, _Float16* __restrict__ out, long total, int E) {
  long i = blockIdx.x * (long)blockDim.x + threadIdx.x;
  if (i >= total) return;
  int ed = (int)(i >> 6);
  int c  = (int)(i & 63);
  int s = ei[ed], d = ei[E + ed];
  float v = (float)xd[(size_t)s * 64 + c] * (float)xd[(size_t)d * 64 + c] * (float)e[i];
  out[i] = (_Float16)v;
}

// Per-edge attention score (4 heads), tanhshrink, atomic segment-sum scatter.
// One wave per edge; lane covers columns {lane, lane+32}.
__global__ void k_scatter(const _Float16* __restrict__ y, const _Float16* __restrict__ e,
                          const _Float16* __restrict__ eaux, const _Float16* __restrict__ m,
                          const int* __restrict__ ei, float* __restrict__ matt, int E) {
  int w    = (int)((blockIdx.x * (long)blockDim.x + threadIdx.x) >> 5);
  int lane = threadIdx.x & 31;
  if (w >= E) return;
  int src = ei[w], dst = ei[E + w];
  size_t eb = (size_t)w * 64;
  float ee0 = (float)e[eb + lane]      * (float)eaux[eb + lane];
  float ee1 = (float)e[eb + lane + 32] * (float)eaux[eb + lane + 32];
  float m0v = (float)m[eb + lane];
  float m1v = (float)m[eb + lane + 32];
  size_t yi = (size_t)dst * 256, yj = (size_t)src * 256;
  float s[4];
#pragma unroll
  for (int h = 0; h < 4; ++h) {
    int c0 = h * 64 + lane, c1 = c0 + 32;
    s[h] = (float)y[yi + c0] * (float)y[yj + c0] * ee0
         + (float)y[yi + c1] * (float)y[yj + c1] * ee1;
  }
#pragma unroll
  for (int off = 16; off > 0; off >>= 1) {
#pragma unroll
    for (int h = 0; h < 4; ++h) s[h] += __shfl_xor(s[h], off, 32);
  }
#pragma unroll
  for (int h = 0; h < 4; ++h) {
    float sm = s[h] * (1.0f / 64.0f);
    float a  = sm - tanhf(sm);                  // tanhshrink
    atomicAdd(&matt[(size_t)dst * 256 + h * 64 + lane],      a * m0v);
    atomicAdd(&matt[(size_t)dst * 256 + h * 64 + lane + 32], a * m1v);
  }
}

// sum_n ( T[n,:] . W2 + b2 ) accumulated into out[0]. One wave per node.
__global__ void k_outsum(const _Float16* __restrict__ T, const float* __restrict__ W2,
                         const float* __restrict__ b2, float* __restrict__ out, int N) {
  int w    = (int)((blockIdx.x * (long)blockDim.x + threadIdx.x) >> 5);
  int lane = threadIdx.x & 31;
  if (w >= N) return;
  size_t base = (size_t)w * 256;
  float s = 0.0f;
#pragma unroll
  for (int i = 0; i < 8; ++i) {
    int c = lane + i * 32;
    s += (float)T[base + c] * W2[c];
  }
#pragma unroll
  for (int off = 16; off > 0; off >>= 1) s += __shfl_xor(s, off, 32);
  if (lane == 0) atomicAdd(out, s + b2[0]);
}

// ---------------------------------------------------------------------------
// Host orchestration
// ---------------------------------------------------------------------------
static void launch_gemm(hipStream_t s, const _Float16* A, int lda, const float* W,
                        int K, int Kpad, const float* bias,
                        _Float16* C, int ldc, int M, int Nc, int act) {
  dim3 grid((M + 63) / 64, Nc / 64);
  k_gemm<<<grid, 128, 0, s>>>(A, lda, W, K, Kpad, bias, C, ldc, M, act);
}

#define EWG(n) ((unsigned)(((n) + 255) / 256))

extern "C" void kernel_launch(void* const* d_in, const int* in_sizes, int n_in,
                              void* d_out, int out_size, void* d_ws, size_t ws_size,
                              hipStream_t stream) {
  (void)in_sizes; (void)n_in; (void)out_size; (void)ws_size;
  const int N = NNODES, E = NEDGES;

  // Inputs (setup_inputs() dict insertion order, recursively flattened)
  const int*   period    = (const int*)d_in[0];
  const int*   group     = (const int*)d_in[1];
  const float* nbf       = (const float*)d_in[2];
  const float* edge_attr = (const float*)d_in[3];
  const int*   eidx      = (const int*)d_in[4];
  const float* embP      = (const float*)d_in[5];
  const float* embG      = (const float*)d_in[6];
  const float* lin_aux   = (const float*)d_in[7];
  const float* enc_e_W   = (const float*)d_in[8];
  const float* enc_e_b   = (const float*)d_in[9];
  const float* enc_e_rW  = (const float*)d_in[10];
  const float* enc_e_rb  = (const float*)d_in[11];
  const float* enc_h_W   = (const float*)d_in[12];
  const float* enc_h_b   = (const float*)d_in[13];
  const float* enc_h_rW  = (const float*)d_in[14];
  const float* enc_h_rb  = (const float*)d_in[15];
  // blocks: base 16 + 15*bi : down_W, down_b, att_W, m_W, m_b, m_g, m_be,
  //                           h_W, h_b, h_g, h_be, e_W, e_b, e_g, e_be
  // outs:   base 61 + 4*oi  : W1, b1, W2, b2

  // Workspace carve-out
  char* p = (char*)d_ws;
  auto carve = [&](size_t bytes) { char* r = p; p += (bytes + 255) & ~(size_t)255; return r; };
  float*    h32    = (float*)   carve((size_t)N * 256 * 4);
  _Float16* h16    = (_Float16*)carve((size_t)N * 256 * 2);
  _Float16* nT0    = (_Float16*)carve((size_t)N * 256 * 2);
  _Float16* nT1    = (_Float16*)carve((size_t)N * 256 * 2);
  _Float16* nEmb   = (_Float16*)carve((size_t)N * 288 * 2);
  _Float16* xdown  = (_Float16*)carve((size_t)N * 64 * 2);
  _Float16* yatt   = (_Float16*)carve((size_t)N * 256 * 2);
  float*    mattF  = (float*)   carve((size_t)N * 256 * 4);
  _Float16* matt16 = (_Float16*)carve((size_t)N * 256 * 2);
  _Float16* eR     = (_Float16*)carve((size_t)E * 32 * 2);
  _Float16* eAux   = (_Float16*)carve((size_t)E * 64 * 2);
  _Float16* eE     = (_Float16*)carve((size_t)E * 64 * 2);
  _Float16* eTa    = (_Float16*)carve((size_t)E * 64 * 2);
  _Float16* eTb    = (_Float16*)carve((size_t)E * 64 * 2);
  _Float16* eM     = (_Float16*)carve((size_t)E * 64 * 2);
  float*    stats  = (float*)   carve(512 * 4);

  float* out = (float*)d_out;
  const long nE64 = (long)E * 64;
  const long nN256 = (long)N * 256;

  k_zero_f32<<<1, 32, 0, stream>>>(out, 1);

  // ---- edge embedding path ----
  k_rbf<<<EWG(E), 256, 0, stream>>>(edge_attr, eR, E);
  launch_gemm(stream, eR, 32, lin_aux, 8, 32, nullptr, eAux, 64, E, 64, 0);
  launch_gemm(stream, eR, 32, enc_e_W, 8, 32, enc_e_b, eE, 64, E, 64, 0);
  launch_gemm(stream, eE,  64, enc_e_rW + 0 * 4096, 64, 64, enc_e_rb + 0,   eTa, 64, E, 64, 1);
  launch_gemm(stream, eTa, 64, enc_e_rW + 1 * 4096, 64, 64, enc_e_rb + 64,  eTb, 64, E, 64, 1);
  launch_gemm(stream, eTb, 64, enc_e_rW + 2 * 4096, 64, 64, enc_e_rb + 128, eTa, 64, E, 64, 1);
  k_add16<<<EWG(nE64), 256, 0, stream>>>(eE, eTa, eE, nE64);

  // ---- node embedding path ----
  k_embed<<<EWG((long)N * 288), 256, 0, stream>>>(period, group, nbf, embP, embG, nEmb, (long)N * 288);
  launch_gemm(stream, nEmb, 288, enc_h_W, 260, 288, enc_h_b, nT0, 256, N, 256, 0);
  launch_gemm(stream, nT0,  256, enc_h_rW + 0 * 65536, 256, 256, enc_h_rb + 0,   nT1,  256, N, 256, 1);
  launch_gemm(stream, nT1,  256, enc_h_rW + 1 * 65536, 256, 256, enc_h_rb + 256, yatt, 256, N, 256, 1);
  launch_gemm(stream, yatt, 256, enc_h_rW + 2 * 65536, 256, 256, enc_h_rb + 512, nT1,  256, N, 256, 1);
  k_hinit<<<EWG(nN256), 256, 0, stream>>>(nT0, nT1, h32, h16, nN256);

  // ---- P = out_block(outs[0], h) ----
  {
    const float* W1 = (const float*)d_in[61]; const float* b1 = (const float*)d_in[62];
    const float* W2 = (const float*)d_in[63]; const float* b2 = (const float*)d_in[64];
    launch_gemm(stream, h16, 256, W1, 256, 256, b1, nT0, 256, N, 256, 1);
    k_outsum<<<(N + 7) / 8, 256, 0, stream>>>(nT0, W2, b2, out, N);
  }

  // ---- interaction blocks ----
  for (int bi = 0; bi < 3; ++bi) {
    const int base = 16 + 15 * bi;
    const float* down_W = (const float*)d_in[base + 0];
    const float* down_b = (const float*)d_in[base + 1];
    const float* att_W  = (const float*)d_in[base + 2];   // [H*EE, EE] = [256,64]
    const float* m_W    = (const float*)d_in[base + 3];
    const float* m_b    = (const float*)d_in[base + 4];
    const float* m_g    = (const float*)d_in[base + 5];
    const float* m_be   = (const float*)d_in[base + 6];
    const float* h_W    = (const float*)d_in[base + 7];
    const float* h_b    = (const float*)d_in[base + 8];
    const float* h_g    = (const float*)d_in[base + 9];
    const float* h_be   = (const float*)d_in[base + 10];
    const float* e_W    = (const float*)d_in[base + 11];
    const float* e_b    = (const float*)d_in[base + 12];
    const float* e_g    = (const float*)d_in[base + 13];
    const float* e_be   = (const float*)d_in[base + 14];

    // x_down = silu(h @ down_W^T + b)   [N,64]
    launch_gemm(stream, h16, 256, down_W, 256, 256, down_b, xdown, 64, N, 64, 1);
    // hoisted per-head projections: yatt[:, h*64+i] = x_down . att_W[h][i]   [N,256]
    launch_gemm(stream, xdown, 64, att_W, 64, 64, nullptr, yatt, 256, N, 256, 0);

    // m = BN+silu( (x_i*x_j*e) @ m_W^T + b )   [E,64]
    k_prod<<<EWG(nE64), 256, 0, stream>>>(xdown, eE, eidx, eTa, nE64, E);
    launch_gemm(stream, eTa, 64, m_W, 64, 64, m_b, eTb, 64, E, 64, 0);
    k_zero_f32<<<2, 256, 0, stream>>>(stats, 512);
    k_colstats<<<1024, 64, 0, stream>>>(eTb, stats, E, 64);
    k_bnsilu<<<EWG(nE64), 256, 0, stream>>>(eTb, stats, m_g, m_be, eM, nE64, 64, 1.0f / (float)E);

    // attention scores + segment_sum scatter into m_att (f32, L2-resident)
    k_zero_f32<<<EWG(nN256), 256, 0, stream>>>(mattF, nN256);
    k_scatter<<<E / 8, 256, 0, stream>>>(yatt, eE, eAux, eM, eidx, mattF, E);

    // e = BN+silu( m @ e_W^T + b )  (after scatter used old e)
    launch_gemm(stream, eM, 64, e_W, 64, 64, e_b, eTa, 64, E, 64, 0);
    k_zero_f32<<<2, 256, 0, stream>>>(stats, 512);
    k_colstats<<<1024, 64, 0, stream>>>(eTa, stats, E, 64);
    k_bnsilu<<<EWG(nE64), 256, 0, stream>>>(eTa, stats, e_g, e_be, eE, nE64, 64, 1.0f / (float)E);

    // h += BN+silu( m_att @ h_W^T + b )
    k_f16cvt<<<EWG(nN256), 256, 0, stream>>>(mattF, matt16, nN256);
    launch_gemm(stream, matt16, 256, h_W, 256, 256, h_b, nT0, 256, N, 256, 0);
    k_zero_f32<<<2, 256, 0, stream>>>(stats, 512);
    k_colstats<<<1024, 256, 0, stream>>>(nT0, stats, N, 256);
    k_bnsilu<<<EWG(nN256), 256, 0, stream>>>(nT0, stats, h_g, h_be, nT1, nN256, 256, 1.0f / (float)N);
    k_hupdate<<<EWG(nN256), 256, 0, stream>>>(nT1, h32, h16, nN256);

    // P += out_block(outs[bi+1], h)
    const int ob = 61 + 4 * (bi + 1);
    const float* W1 = (const float*)d_in[ob + 0]; const float* b1 = (const float*)d_in[ob + 1];
    const float* W2 = (const float*)d_in[ob + 2]; const float* b2 = (const float*)d_in[ob + 3];
    launch_gemm(stream, h16, 256, W1, 256, 256, b1, nT0, 256, N, 256, 1);
    k_outsum<<<(N + 7) / 8, 256, 0, stream>>>(nT0, W2, b2, out, N);
  }
}
